// Attention_47450798686504
// MI455X (gfx1250) — matrix operand order repaired and checked by reading, not compiled, not run on hardware
//
#include <hip/hip_runtime.h>

typedef __bf16 bf16;
typedef __attribute__((ext_vector_type(16))) __bf16 v16bf;
typedef __attribute__((ext_vector_type(8)))  float  v8f;

#define T_SEQ 2048
#define DM    1024
#define NH    16
#define HD    64

#define WMMA_BF16(A,B,C) \
  __builtin_amdgcn_wmma_f32_16x16x32_bf16(false,(A),false,(B),(short)0,(C),false,false)

// A-matrix 16x32 bf16 fragment from LDS tile (row-major, 'stride' elements per row).
// lane (l, hi): holds row M=l; VGPR j holds K = 2j + (j>=4 ? 8:0) + 8*hi (pairs).
static __device__ inline v16bf frag_a_lds(const bf16* base, int stride, int l, int hi) {
  union { v16bf v; unsigned u[8]; } r;
  const unsigned short* row = (const unsigned short*)base + (size_t)l * stride;
#pragma unroll
  for (int j = 0; j < 8; ++j) {
    int k = 2*j + ((j >= 4) ? 8 : 0) + 8*hi;
    r.u[j] = *(const unsigned*)(row + k);
  }
  return r.v;
}

// B-matrix 32x16 bf16 fragment where tile stores B transposed: tile[n][k].
static __device__ inline v16bf frag_b_rows(const bf16* tile, int stride, int row, int kbase) {
  union { v16bf v; unsigned u[8]; } r;
  const unsigned short* p = (const unsigned short*)tile + (size_t)row * stride + kbase;
#pragma unroll
  for (int j = 0; j < 8; ++j) r.u[j] = *(const unsigned*)(p + 2*j + ((j >= 4) ? 8 : 0));
  return r.v;
}

// C = A * W^T.  A: [8192, 1024] (fp32 or bf16).  W: [1024(out), 1024(in)] fp32.
// Block tile 64M x 128N, 128 threads (4 waves), wave tile 32x64 (8 WMMA / k-chunk).
// scatter=1: write bf16 to [B,H,T,hd].  scatter=0: write fp32 row-major.
__global__ __launch_bounds__(128)
void gemm_xwt_kernel(const float* __restrict__ Af, const bf16* __restrict__ Ab,
                     const float* __restrict__ W,
                     bf16* __restrict__ out_sc, float* __restrict__ out_f,
                     int a_is_f32, int scatter)
{
  __shared__ __align__(16) bf16 At[64][34];
  __shared__ __align__(16) bf16 Wt[128][34];
  const int tid = threadIdx.x;
  const int w  = __builtin_amdgcn_readfirstlane(tid >> 5);
  const int l  = tid & 15, hi = (tid >> 4) & 1;
  const int m0 = blockIdx.x * 64, n0 = blockIdx.y * 128;
  const int mo = 32 * (w >> 1), no = 64 * (w & 1);

  const int ar  = tid >> 3, ac  = (tid & 7) * 4;   // f32 tiles: row, col(float)
  const int ar2 = tid >> 2, ac2 = (tid & 3) * 8;   // bf16 A: row, col(elem)
  size_t aoff_f = (size_t)(m0 + ar) * DM + ac;            // float index
  size_t aoff_h = ((size_t)(m0 + ar2) * DM + ac2) >> 3;   // uint4 index
  size_t woff   = (size_t)(n0 + ar) * DM + ac;
  const uint4* Ab4 = (const uint4*)Ab;

  v8f acc[2][4] = {};

  for (int kc = 0; kc < DM / 32; ++kc) {
    // prefetch next tiles into registers (overlaps previous chunk's WMMAs)
    float4 fa[4]; uint4 ha[2]; float4 fw[8];
    if (a_is_f32) {
#pragma unroll
      for (int e = 0; e < 4; ++e)
        fa[e] = *(const float4*)(Af + aoff_f + (size_t)e * 16 * DM);
    } else {
#pragma unroll
      for (int e = 0; e < 2; ++e)
        ha[e] = Ab4[aoff_h + (size_t)e * 32 * (DM / 8)];
    }
#pragma unroll
    for (int e = 0; e < 8; ++e)
      fw[e] = *(const float4*)(W + woff + (size_t)e * 16 * DM);
    aoff_f += 32; aoff_h += 4; woff += 32;

    __syncthreads();
    if (a_is_f32) {
#pragma unroll
      for (int e = 0; e < 4; ++e) {
        union { unsigned u[2]; bf16 h[4]; } pk;
        pk.h[0] = (bf16)fa[e].x; pk.h[1] = (bf16)fa[e].y;
        pk.h[2] = (bf16)fa[e].z; pk.h[3] = (bf16)fa[e].w;
        *(unsigned*)&At[ar + 16 * e][ac]     = pk.u[0];
        *(unsigned*)&At[ar + 16 * e][ac + 2] = pk.u[1];
      }
    } else {
#pragma unroll
      for (int e = 0; e < 2; ++e) {
        *(unsigned*)&At[ar2 + 32 * e][ac2]     = ha[e].x;
        *(unsigned*)&At[ar2 + 32 * e][ac2 + 2] = ha[e].y;
        *(unsigned*)&At[ar2 + 32 * e][ac2 + 4] = ha[e].z;
        *(unsigned*)&At[ar2 + 32 * e][ac2 + 6] = ha[e].w;
      }
    }
#pragma unroll
    for (int e = 0; e < 8; ++e) {
      union { unsigned u[2]; bf16 h[4]; } pk;
      pk.h[0] = (bf16)fw[e].x; pk.h[1] = (bf16)fw[e].y;
      pk.h[2] = (bf16)fw[e].z; pk.h[3] = (bf16)fw[e].w;
      *(unsigned*)&Wt[ar + 16 * e][ac]     = pk.u[0];
      *(unsigned*)&Wt[ar + 16 * e][ac + 2] = pk.u[1];
    }
    __syncthreads();

    v16bf a0 = frag_a_lds(&At[mo][0],      34, l, hi);
    v16bf a1 = frag_a_lds(&At[mo + 16][0], 34, l, hi);
#pragma unroll
    for (int ni = 0; ni < 4; ++ni) {
      v16bf b = frag_b_rows(&Wt[no + 16 * ni][0], 34, l, 8 * hi);
      acc[0][ni] = WMMA_BF16(a0, b, acc[0][ni]);
      acc[1][ni] = WMMA_BF16(a1, b, acc[1][ni]);
    }
  }

#pragma unroll
  for (int mi = 0; mi < 2; ++mi)
#pragma unroll
    for (int ni = 0; ni < 4; ++ni)
#pragma unroll
      for (int r = 0; r < 8; ++r) {
        int m = m0 + mo + 16 * mi + r + 8 * hi;
        int n = n0 + no + 16 * ni + l;
        float v = acc[mi][ni][r];
        if (scatter) {
          int b = m >> 11, t = m & (T_SEQ - 1);
          int h = n >> 6,  d = n & 63;
          out_sc[(((size_t)(b * NH + h)) * T_SEQ + t) * HD + d] = (bf16)v;
        } else {
          out_f[(size_t)m * DM + n] = v;
        }
      }
}

// Flash attention: one block per (64-query tile, head). Q/K/V bf16 in [B,H,T,hd].
// Output: bf16 [B, T, H*hd] (heads concatenated) for the final projection.
__global__ __launch_bounds__(128)
void attn_kernel(const bf16* __restrict__ Q, const bf16* __restrict__ K,
                 const bf16* __restrict__ V, bf16* __restrict__ O)
{
  __shared__ __align__(16) bf16 Kt[32][66];    // [key][d]
  __shared__ __align__(16) bf16 VtT[64][34];   // [d][key]  (transposed at load)
  __shared__ __align__(16) bf16 Pt[4][16][34];

  const int tid = threadIdx.x;
  const int w  = __builtin_amdgcn_readfirstlane(tid >> 5);
  const int l  = tid & 15;
  const int hi = (tid >> 4) & 1;
  const int q0 = blockIdx.x * 64;
  const int bh = blockIdx.y;
  const int b = bh >> 4, h = bh & 15;
  const size_t head = (size_t)bh * T_SEQ * HD;  // element offset of this head
  const int qrow0 = q0 + 16 * w;

  // Q fragments for this wave's 16 rows (two 16x32 chunks over hd=64).
  v16bf qf[2];
  {
    const unsigned* qrow = (const unsigned*)((const unsigned short*)Q + head)
                           + (size_t)(qrow0 + l) * (HD / 2);
#pragma unroll
    for (int c = 0; c < 2; ++c) {
      union { v16bf v; unsigned u[8]; } r;
#pragma unroll
      for (int j = 0; j < 8; ++j) {
        int k = 32 * c + 2 * j + ((j >= 4) ? 8 : 0) + 8 * hi;
        r.u[j] = qrow[k >> 1];
      }
      qf[c] = r.v;
    }
  }

  v8f acc[4] = {};
  float mrow[8], lrow[8];
#pragma unroll
  for (int r = 0; r < 8; ++r) { mrow[r] = -3.0e38f; lrow[r] = 0.f; }

  const uint4* K4 = (const uint4*)K;
  const uint4* V4 = (const uint4*)V;
  const int lr = tid >> 3, lc = (tid & 7) * 8;              // tile row(key), col(d)
  size_t kvoff = (head >> 3) + (size_t)lr * (HD / 8) + (tid & 7);  // uint4 index

  const int nkt = 2 * blockIdx.x + 2;  // causal: key tiles up to q0+63
  for (int kt = 0; kt < nkt; ++kt) {
    const int k0 = kt * 32;
    uint4 ku[2], vu[2];
#pragma unroll
    for (int e = 0; e < 2; ++e) {
      ku[e] = K4[kvoff + (size_t)e * 16 * (HD / 8)];
      vu[e] = V4[kvoff + (size_t)e * 16 * (HD / 8)];
    }
    kvoff += 32 * (HD / 8);

    __syncthreads();
#pragma unroll
    for (int e = 0; e < 2; ++e) {
      *(unsigned*)&Kt[lr + 16 * e][lc]     = ku[e].x;
      *(unsigned*)&Kt[lr + 16 * e][lc + 2] = ku[e].y;
      *(unsigned*)&Kt[lr + 16 * e][lc + 4] = ku[e].z;
      *(unsigned*)&Kt[lr + 16 * e][lc + 6] = ku[e].w;
      // V stored transposed: VtT[d][key] so P*V B-fragments are packed pairs
      union { uint4 q; bf16 hh[8]; } vv; vv.q = vu[e];
#pragma unroll
      for (int i = 0; i < 8; ++i)
        VtT[lc + i][lr + 16 * e] = vv.hh[i];
    }
    __syncthreads();

    if (k0 <= qrow0 + 15) {      // wave-uniform: tile intersects this strip
      v8f s0 = {}, s1 = {};
#pragma unroll
      for (int c = 0; c < 2; ++c) {  // d-chunks of 32
        v16bf bk0 = frag_b_rows(&Kt[0][0],  66, l, 32 * c + 8 * hi);
        v16bf bk1 = frag_b_rows(&Kt[16][0], 66, l, 32 * c + 8 * hi);
        s0 = WMMA_BF16(qf[c], bk0, s0);
        s1 = WMMA_BF16(qf[c], bk1, s1);
      }

      float scal[8];
#pragma unroll
      for (int r = 0; r < 8; ++r) {
        int q    = qrow0 + r + 8 * hi;
        int key0 = k0 + l, key1 = k0 + 16 + l;
        bool ok0 = key0 <= q, ok1 = key1 <= q;
        float v0 = ok0 ? s0[r] * 0.125f : -3.0e38f;  // 1/sqrt(64)
        float v1 = ok1 ? s1[r] * 0.125f : -3.0e38f;
        float mx = fmaxf(v0, v1);
#pragma unroll
        for (int msk = 1; msk <= 8; msk <<= 1) mx = fmaxf(mx, __shfl_xor(mx, msk, 32));
        float mn = fmaxf(mrow[r], mx);
        float sc = __expf(mrow[r] - mn);
        float e0 = ok0 ? __expf(v0 - mn) : 0.f;
        float e1 = ok1 ? __expf(v1 - mn) : 0.f;
        float rs = e0 + e1;
#pragma unroll
        for (int msk = 1; msk <= 8; msk <<= 1) rs += __shfl_xor(rs, msk, 32);
        mrow[r] = mn;
        lrow[r] = lrow[r] * sc + rs;
        scal[r] = sc;
        Pt[w][r + 8 * hi][l]      = (bf16)e0;   // C-layout -> LDS
        Pt[w][r + 8 * hi][16 + l] = (bf16)e1;
      }
#pragma unroll
      for (int t = 0; t < 4; ++t)
#pragma unroll
        for (int r = 0; r < 8; ++r) acc[t][r] *= scal[r];

      // LDS -> A-layout (same-wave DS ops are in-order; no barrier needed)
      v16bf pf = frag_a_lds(&Pt[w][0][0], 34, l, hi);
#pragma unroll
      for (int t = 0; t < 4; ++t) {
        // B[key, d] from VtT[d][key]: packed key-pairs per lane
        v16bf bv = frag_b_rows(&VtT[0][0], 34, 16 * t + l, 8 * hi);
        acc[t] = WMMA_BF16(pf, bv, acc[t]);
      }
    }
  }

#pragma unroll
  for (int t = 0; t < 4; ++t)
#pragma unroll
    for (int r = 0; r < 8; ++r) {
      int qq = qrow0 + r + 8 * hi;
      float o = acc[t][r] / lrow[r];
      O[((size_t)b * T_SEQ + qq) * DM + h * HD + 16 * t + l] = (bf16)o;
    }
}

extern "C" void kernel_launch(void* const* d_in, const int* in_sizes, int n_in,
                              void* d_out, int out_size, void* d_ws, size_t ws_size,
                              hipStream_t stream) {
  const float* x  = (const float*)d_in[0];
  const float* Wq = (const float*)d_in[1];
  const float* Wk = (const float*)d_in[2];
  const float* Wv = (const float*)d_in[3];
  const float* Wo = (const float*)d_in[4];
  float* out = (float*)d_out;

  const size_t M = 4 * (size_t)T_SEQ;        // 8192
  bf16* qb = (bf16*)d_ws;
  bf16* kb = qb + M * DM;
  bf16* vb = kb + M * DM;
  bf16* ob = vb + M * DM;

  dim3 ggrid(M / 64, DM / 128);              // 128 x 8
  gemm_xwt_kernel<<<ggrid, 128, 0, stream>>>(x, nullptr, Wq, qb, nullptr, 1, 1);
  gemm_xwt_kernel<<<ggrid, 128, 0, stream>>>(x, nullptr, Wk, kb, nullptr, 1, 1);
  gemm_xwt_kernel<<<ggrid, 128, 0, stream>>>(x, nullptr, Wv, vb, nullptr, 1, 1);

  dim3 agrid(T_SEQ / 64, 4 * NH);            // 32 x 64
  attn_kernel<<<agrid, 128, 0, stream>>>(qb, kb, vb, ob);

  gemm_xwt_kernel<<<ggrid, 128, 0, stream>>>(nullptr, ob, Wo, nullptr, out, 0, 0);
}
